// wordEncoder_76501957476684
// MI455X (gfx1250) — compile-verified, matrix operand order checked by reading, not run on hardware
//
#include <hip/hip_runtime.h>
#include <hip/hip_bf16.h>
#include <math.h>
#include <stdint.h>

typedef __attribute__((ext_vector_type(16))) _Float16 v16h;
typedef __attribute__((ext_vector_type(8)))  float    v8f;
typedef __attribute__((__vector_size__(4 * sizeof(int)))) int v4i_t;

#define E   128
#define H   128
#define H3  384
#define O   128
#define BB  64
#define TT  512

// ---- CDNA5 async LDS copy (ASYNCcnt path) ----------------------------------
#if defined(__has_builtin)
#if __has_builtin(__builtin_amdgcn_global_load_async_to_lds_b128)
#define ASYNC_LDS 1
#endif
#endif

__device__ __forceinline__ void async_copy16(const _Float16* g, _Float16* l) {
#ifdef ASYNC_LDS
  __builtin_amdgcn_global_load_async_to_lds_b128((v4i_t*)(void*)g, (v4i_t*)l, 0, 0);
#else
  *(float4*)l = *(const float4*)g;
#endif
}

__device__ __forceinline__ void wait_async0() {
#ifdef ASYNC_LDS
#if __has_builtin(__builtin_amdgcn_s_wait_asynccnt)
  __builtin_amdgcn_s_wait_asynccnt(0);
#else
  asm volatile("s_wait_asynccnt 0x0" ::: "memory");
#endif
#endif
}

// ---- WMMA fragment helpers (ISA 7.12.2 layouts, wave32) ---------------------
// 16-bit A/B fragment, K-chunk of 32:
//   lane<16 : K = 0..7 (v0..v3) and 16..23 (v4..v7)
//   lane>=16: K = 8..15          and 24..31
__device__ __forceinline__ int klocal(int lane, int i) {
  return ((i & 8) << 1) + ((lane & 16) >> 1) + (i & 7);
}

// Load a 16x32 fragment from a row-major matrix (works for A rows and for
// B = W[n][k] row-major, since the B fragment wants column n of B == row n of W).
__device__ __forceinline__ v16h frag_rm(const _Float16* base, int ld, int k0, int lane) {
  v16h f;
  const _Float16* p = base + (lane & 15) * ld + k0;
#pragma unroll
  for (int i = 0; i < 16; ++i) f[i] = p[klocal(lane, i)];
  return f;
}

__device__ __forceinline__ v8f wmma_f16(v16h a, v16h b, v8f c) {
  return __builtin_amdgcn_wmma_f32_16x16x32_f16(false, a, false, b, (short)0, c,
                                                false, false);
}

// ---- K0: weight cast f32 -> f16 --------------------------------------------
__global__ void k_cast(const float* __restrict__ src, _Float16* __restrict__ dst, int n) {
  int i = blockIdx.x * blockDim.x + threadIdx.x;
  if (i < n) dst[i] = (_Float16)src[i];
}

// ---- K1: fused embedding + e2i projection + input-gate precompute ----------
// grid: T * (B/16) blocks of 256 threads (8 waves).  One block = 16 tokens
// (fixed t, 16 consecutive b).  gx layout: [T][B][3H] f16, per direction.
__global__ __launch_bounds__(256) void k_gates(
    const int* __restrict__ Xtok, const float* __restrict__ emb,
    const _Float16* __restrict__ We2i, const float* __restrict__ be2i,
    const _Float16* __restrict__ Wihf, const float* __restrict__ bihf,
    const _Float16* __restrict__ Wihb, const float* __restrict__ bihb,
    _Float16* __restrict__ gxf, _Float16* __restrict__ gxb) {
  __shared__ _Float16 eh[16 * E];
  __shared__ _Float16 xh[16 * H];
  __shared__ int stok[16];
  const int g = blockIdx.x;
  const int t = g >> 2;
  const int b0 = (g & 3) * 16;
  const int tid = threadIdx.x;
  const int lane = tid & 31;
  const int w = tid >> 5;

  if (tid < 16) stok[tid] = Xtok[(b0 + tid) * TT + t];
  __syncthreads();

  // gather 16 embedding rows -> LDS (f16)
  for (int i = tid; i < 16 * E; i += 256) {
    int m = i >> 7, k = i & 127;
    eh[i] = (_Float16)emb[(size_t)stok[m] * E + k];
  }
  __syncthreads();

  // xh = eh @ We2i^T + be2i   (8 n-tiles, one per wave)
  {
    const int n0 = w * 16;
    v8f c = {};
#pragma unroll
    for (int kc = 0; kc < E; kc += 32)
      c = wmma_f16(frag_rm(eh, E, kc, lane), frag_rm(We2i + n0 * E, E, kc, lane), c);
    const int n = lane & 15, mh = (lane >> 4) << 3;
#pragma unroll
    for (int r = 0; r < 8; ++r)
      xh[(mh + r) * H + n0 + n] = (_Float16)(c[r] + be2i[n0 + n]);
  }
  __syncthreads();

  // gx_{f,b} = xh @ Wih^T + bih   (24 n-tiles x 2 dirs, 6 per wave, uniform)
  for (int tt = w; tt < 48; tt += 8) {
    const int dir = tt / 24;
    const int nt = tt % 24;
    const int n0 = nt * 16;
    const _Float16* Wih = dir ? Wihb : Wihf;
    const float* bih = dir ? bihb : bihf;
    _Float16* gx = dir ? gxb : gxf;
    v8f c = {};
#pragma unroll
    for (int kc = 0; kc < H; kc += 32)
      c = wmma_f16(frag_rm(xh, H, kc, lane), frag_rm(Wih + n0 * H, H, kc, lane), c);
    const int n = lane & 15, mh = (lane >> 4) << 3;
#pragma unroll
    for (int r = 0; r < 8; ++r)
      gx[((size_t)t * BB + b0 + mh + r) * H3 + n0 + n] = (_Float16)(c[r] + bih[n0 + n]);
  }
}

// ---- K2: sequential GRU recurrence, grid=2 (dir = blockIdx.x) --------------
// 1024 threads = 32 waves.  Dynamic LDS (304KB < 320KB WGP):
//   Whh 96KB f16 | h 16KB f16 | gh 96KB f32 | gx double-buffer 2x48KB f16
// Per step: async-prefetch gx[t+1] into LDS (ASYNCcnt) while doing 96 WMMA
// tiles (4 m-tiles x 24 n-tiles, 3 per wave, K=128) and the gate elementwise.
#define LDS_WHH 0
#define LDS_H   (H3 * H * 2)
#define LDS_GH  (LDS_H + BB * H * 2)
#define LDS_GX  (LDS_GH + BB * H3 * 4)
#define LDS_TOT (LDS_GX + 2 * BB * H3 * 2)

__global__ __launch_bounds__(1024) void k_gru(
    const _Float16* __restrict__ gxf, const _Float16* __restrict__ gxb,
    const _Float16* __restrict__ Whhf_h, const _Float16* __restrict__ Whhb_h,
    const float* __restrict__ bhhf, const float* __restrict__ bhhb,
    const int* __restrict__ lens, _Float16* __restrict__ hbuf) {
  extern __shared__ char smem[];
  _Float16* sWhh = (_Float16*)(smem + LDS_WHH);  // [3H][H] f16
  _Float16* sh   = (_Float16*)(smem + LDS_H);    // [B][H] f16
  float*    sgh  = (float*)(smem + LDS_GH);      // [B][3H] f32
  _Float16* sgx  = (_Float16*)(smem + LDS_GX);   // [2][B][3H] f16
  __shared__ float sbhh[H3];
  __shared__ int   slens[BB];

  const int dir = blockIdx.x;
  const _Float16* gx = dir ? gxb : gxf;
  const _Float16* Whh = dir ? Whhb_h : Whhf_h;
  const float* bhh = dir ? bhhb : bhhf;
  const int tid = threadIdx.x;
  const int lane = tid & 31;
  const int w = tid >> 5;

  for (int i = tid; i < H3 * H; i += 1024) sWhh[i] = Whh[i];
  for (int i = tid; i < BB * H; i += 1024) sh[i] = (_Float16)0.0f;
  if (tid < H3) sbhh[tid] = bhh[tid];
  if (tid < BB) slens[tid] = lens[tid];

  // preload gx[t0] into buffer 0 (48KB = 3072 x 16B, 3 per thread)
  {
    const int t0 = dir ? (TT - 1) : 0;
    const _Float16* src = gx + (size_t)t0 * BB * H3;
#pragma unroll
    for (int k = 0; k < 3; ++k) {
      const int idx = tid + k * 1024;
      async_copy16(src + idx * 8, sgx + idx * 8);
    }
    wait_async0();
  }
  __syncthreads();

  for (int s = 0; s < TT; ++s) {
    const int t = dir ? (TT - 1 - s) : s;
    const _Float16* gxt = sgx + (s & 1) * (BB * H3);

    // kick off async prefetch of next step's gates into the other buffer
    if (s + 1 < TT) {
      const int tn = dir ? (t - 1) : (t + 1);
      const _Float16* src = gx + (size_t)tn * BB * H3;
      _Float16* dst = sgx + ((s + 1) & 1) * (BB * H3);
#pragma unroll
      for (int k = 0; k < 3; ++k) {
        const int idx = tid + k * 1024;
        async_copy16(src + idx * 8, dst + idx * 8);
      }
    }

    // gh = h @ Whh^T   (tile = nt*4 + mt)
#pragma unroll
    for (int it = 0; it < 3; ++it) {
      const int tile = w + it * 32;
      const int m0 = (tile & 3) * 16;
      const int n0 = (tile >> 2) * 16;
      v8f c = {};
#pragma unroll
      for (int kc = 0; kc < H; kc += 32)
        c = wmma_f16(frag_rm(sh + m0 * H, H, kc, lane),
                     frag_rm(sWhh + n0 * H, H, kc, lane), c);
      const int n = lane & 15, mh = (lane >> 4) << 3;
#pragma unroll
      for (int r = 0; r < 8; ++r)
        sgh[(m0 + mh + r) * H3 + n0 + n] = c[r];
    }
    __syncthreads();

    // elementwise GRU cell: 64*128 elems, 8 per thread (gates read from LDS)
#pragma unroll
    for (int e = 0; e < 8; ++e) {
      const int idx = tid + e * 1024;
      const int b = idx >> 7;
      const int j = idx & 127;
      const float gxr = (float)gxt[b * H3 + j];
      const float gxz = (float)gxt[b * H3 + 128 + j];
      const float gxn = (float)gxt[b * H3 + 256 + j];
      const float ghr = sgh[b * H3 + j] + sbhh[j];
      const float ghz = sgh[b * H3 + 128 + j] + sbhh[128 + j];
      const float ghn = sgh[b * H3 + 256 + j] + sbhh[256 + j];
      const float r = 1.0f / (1.0f + __expf(-(gxr + ghr)));
      const float z = 1.0f / (1.0f + __expf(-(gxz + ghz)));
      const float nn = tanhf(gxn + r * ghn);
      const float hold = (float)sh[b * H + j];
      const float hnew = (1.0f - z) * nn + z * hold;
      const bool valid = t < slens[b];
      sh[b * H + j] = (_Float16)(valid ? hnew : hold);
      hbuf[((size_t)t * BB + b) * (2 * H) + dir * H + j] =
          (_Float16)(valid ? hnew : 0.0f);
    }
    // make sure next step's gate buffer has landed, then step barrier
    wait_async0();
    __syncthreads();
  }
}

// ---- K3: Y = tanh(Hout@Wh2o^T + b), score = sum(Y*u) -----------------------
__global__ __launch_bounds__(256) void k_scores(
    const _Float16* __restrict__ hbuf, const _Float16* __restrict__ Wh2o_h,
    const float* __restrict__ bh2o, const float* __restrict__ uw,
    float* __restrict__ scores) {
  __shared__ float wsc[8][16];
  const int g = blockIdx.x;
  const int t = g >> 2;
  const int b0 = (g & 3) * 16;
  const int tid = threadIdx.x, lane = tid & 31, w = tid >> 5;
  const int n0 = w * 16;
  const _Float16* A = hbuf + ((size_t)t * BB + b0) * (2 * H);

  v8f c = {};
#pragma unroll
  for (int kc = 0; kc < 2 * H; kc += 32)
    c = wmma_f16(frag_rm(A, 2 * H, kc, lane),
                 frag_rm(Wh2o_h + n0 * (2 * H), 2 * H, kc, lane), c);

  const int n = lane & 15, mh = (lane >> 4) << 3;
  float part[8];
#pragma unroll
  for (int r = 0; r < 8; ++r)
    part[r] = tanhf(c[r] + bh2o[n0 + n]) * uw[n0 + n];

  // reduce over the 16 lanes that share the same rows (n = 0..15)
#pragma unroll
  for (int off = 8; off >= 1; off >>= 1)
#pragma unroll
    for (int r = 0; r < 8; ++r) part[r] += __shfl_xor(part[r], off, 32);

  if ((lane & 15) == 0)
#pragma unroll
    for (int r = 0; r < 8; ++r) wsc[w][mh + r] = part[r];
  __syncthreads();

  if (tid < 16) {
    float sacc = 0.0f;
#pragma unroll
    for (int ww = 0; ww < 8; ++ww) sacc += wsc[ww][tid];
    scores[(size_t)(b0 + tid) * TT + t] = sacc;  // scores[B][T]
  }
}

// ---- K4: masked softmax over t + pooled = sum_t alpha * Hout ---------------
__global__ __launch_bounds__(256) void k_pool(
    const float* __restrict__ scores, const _Float16* __restrict__ hbuf,
    const int* __restrict__ lens, float* __restrict__ out) {
  __shared__ float red[256];
  __shared__ float s_max, s_sum;
  const int b = blockIdx.x;
  const int tid = threadIdx.x;
  const int len = lens[b];

  float mx = -1e30f;
  for (int t = tid; t < len; t += 256) mx = fmaxf(mx, scores[(size_t)b * TT + t]);
  red[tid] = mx;
  __syncthreads();
  for (int s = 128; s >= 1; s >>= 1) {
    if (tid < s) red[tid] = fmaxf(red[tid], red[tid + s]);
    __syncthreads();
  }
  if (tid == 0) s_max = red[0];
  __syncthreads();
  const float mxv = s_max;

  float sm = 0.0f;
  for (int t = tid; t < len; t += 256) sm += __expf(scores[(size_t)b * TT + t] - mxv);
  red[tid] = sm;
  __syncthreads();
  for (int s = 128; s >= 1; s >>= 1) {
    if (tid < s) red[tid] += red[tid + s];
    __syncthreads();
  }
  if (tid == 0) s_sum = red[0];
  __syncthreads();
  const float inv = 1.0f / s_sum;

  const int d = tid;  // 0..255 output feature
  float acc = 0.0f;
  for (int t = 0; t < len; ++t) {
    const float a = __expf(scores[(size_t)b * TT + t] - mxv) * inv;
    acc += a * (float)hbuf[((size_t)t * BB + b) * (2 * H) + d];
  }
  out[b * (2 * H) + d] = acc;  // [1,B,2H] flattened
}

// ---------------------------------------------------------------------------
extern "C" void kernel_launch(void* const* d_in, const int* in_sizes, int n_in,
                              void* d_out, int out_size, void* d_ws, size_t ws_size,
                              hipStream_t stream) {
  const int*   X    = (const int*)d_in[0];
  const int*   lens = (const int*)d_in[1];
  const float* emb  = (const float*)d_in[3];
  const float* We2i = (const float*)d_in[4];
  const float* be2i = (const float*)d_in[5];
  const float* Wihf = (const float*)d_in[6];
  const float* Whhf = (const float*)d_in[7];
  const float* bihf = (const float*)d_in[8];
  const float* bhhf = (const float*)d_in[9];
  const float* Wihb = (const float*)d_in[10];
  const float* Whhb = (const float*)d_in[11];
  const float* bihb = (const float*)d_in[12];
  const float* bhhb = (const float*)d_in[13];
  const float* Wh2o = (const float*)d_in[14];
  const float* bh2o = (const float*)d_in[15];
  const float* uw   = (const float*)d_in[16];
  float* out = (float*)d_out;

  char* ws = (char*)d_ws;
  size_t off = 0;
  auto alloc = [&](size_t bytes) -> void* {
    void* p = ws + off;
    off = (off + bytes + 255) & ~(size_t)255;
    return p;
  };
  _Float16* hWe2i = (_Float16*)alloc((size_t)H * E * 2);
  _Float16* hWihf = (_Float16*)alloc((size_t)H3 * H * 2);
  _Float16* hWhhf = (_Float16*)alloc((size_t)H3 * H * 2);
  _Float16* hWihb = (_Float16*)alloc((size_t)H3 * H * 2);
  _Float16* hWhhb = (_Float16*)alloc((size_t)H3 * H * 2);
  _Float16* hWh2o = (_Float16*)alloc((size_t)O * 2 * H * 2);
  _Float16* gxf   = (_Float16*)alloc((size_t)TT * BB * H3 * 2);
  _Float16* gxb   = (_Float16*)alloc((size_t)TT * BB * H3 * 2);
  _Float16* hbuf  = (_Float16*)alloc((size_t)TT * BB * 2 * H * 2);
  float*    scr   = (float*)alloc((size_t)BB * TT * 4);

  // K0: weight casts
  k_cast<<<(H * E + 255) / 256, 256, 0, stream>>>(We2i, hWe2i, H * E);
  k_cast<<<(H3 * H + 255) / 256, 256, 0, stream>>>(Wihf, hWihf, H3 * H);
  k_cast<<<(H3 * H + 255) / 256, 256, 0, stream>>>(Whhf, hWhhf, H3 * H);
  k_cast<<<(H3 * H + 255) / 256, 256, 0, stream>>>(Wihb, hWihb, H3 * H);
  k_cast<<<(H3 * H + 255) / 256, 256, 0, stream>>>(Whhb, hWhhb, H3 * H);
  k_cast<<<(O * 2 * H + 255) / 256, 256, 0, stream>>>(Wh2o, hWh2o, O * 2 * H);

  // K1: embed + input projection + gate precompute (WMMA)
  k_gates<<<TT * (BB / 16), 256, 0, stream>>>(X, emb, hWe2i, be2i, hWihf, bihf,
                                              hWihb, bihb, gxf, gxb);

  // K2: sequential GRU, both directions concurrently (grid=2)
  (void)hipFuncSetAttribute((const void*)k_gru,
                            hipFuncAttributeMaxDynamicSharedMemorySize, LDS_TOT);
  k_gru<<<2, 1024, LDS_TOT, stream>>>(gxf, gxb, hWhhf, hWhhb, bhhf, bhhb, lens, hbuf);

  // K3: output projection + attention scores (WMMA)
  k_scores<<<TT * (BB / 16), 256, 0, stream>>>(hbuf, hWh2o, bh2o, uw, scr);

  // K4: per-sample masked softmax + pooled output
  k_pool<<<BB, 256, 0, stream>>>(scr, hbuf, lens, out);
}
